// GNN_MetaLayer_Model_25134148616722
// MI455X (gfx1250) — matrix operand order repaired and checked by reading, not compiled, not run on hardware
//
#include <hip/hip_runtime.h>
#include <hip/hip_bf16.h>

#define DEV __device__ __forceinline__

typedef __attribute__((ext_vector_type(16))) __bf16 bf16x16;
typedef __attribute__((ext_vector_type(8)))  __bf16 bf16x8;
typedef __attribute__((ext_vector_type(4)))  __bf16 bf16x4;
typedef __attribute__((ext_vector_type(8)))  float  f32x8;
typedef __attribute__((ext_vector_type(4)))  float  f32x4;

static constexpr int NODES    = 50000;
static constexpr int EDGES    = 800000;
static constexpr int D        = 64;      // node/edge feature dim
static constexpr int TM       = 128;     // rows per block tile
static constexpr int NTHREADS = 256;     // 8 wave32 waves
static constexpr int LD_E1    = 3 * D + 8;  // 200: padded LDS stride for [*,192]
static constexpr int LD_128   = 2 * D + 8;  // 136: padded LDS stride for [*,128]

DEV void atomic_add_f32(float* p, float v) {
  __hip_atomic_fetch_add(p, v, __ATOMIC_RELAXED, __HIP_MEMORY_SCOPE_AGENT);
}

DEV bf16x16 ld_frag16(const __bf16* p0, const __bf16* p1) {
  bf16x8 lo = *reinterpret_cast<const bf16x8*>(p0);
  bf16x8 hi = *reinterpret_cast<const bf16x8*>(p1);
  return __builtin_shufflevector(lo, hi, 0, 1, 2, 3, 4, 5, 6, 7,
                                         8, 9, 10, 11, 12, 13, 14, 15);
}

// One wave computes a 16 x (NT*16) block of  act[128,K] @ W[K, NT*16].
// act: LDS bf16, row-major, leading dim ldA (multiple of 8).
// wT : LDS bf16, W stored TRANSPOSED [Nout, K], leading dim ldB (multiple of 8).
// A fragment (16x32 bf16): lane l<16 holds row M=l, K = {ko..ko+7, ko+16..ko+23};
//                          lane l>=16 holds K = {ko+8..ko+15, ko+24..ko+31}.
// B fragment (32x16 bf16): lane l<16 holds col N=l, K = {ko..ko+7, ko+16..ko+23};
//                          lane l>=16 holds K = {ko+8..ko+15, ko+24..ko+31}.
template <int K, int NT>
DEV void wave_gemm(const __bf16* act, int ldA, const __bf16* wT, int ldB,
                   int mRow, int lane, f32x8* acc) {
#pragma unroll
  for (int nt = 0; nt < NT; ++nt) {
#pragma unroll
    for (int j = 0; j < 8; ++j) acc[nt][j] = 0.0f;
  }
  const int l15 = lane & 15;
  const int hi  = (lane >> 4) & 1;
#pragma unroll
  for (int ko = 0; ko < K; ko += 32) {
    const __bf16* arow = act + (mRow + l15) * ldA + ko + hi * 8;
    bf16x16 a = ld_frag16(arow, arow + 16);
#pragma unroll
    for (int nt = 0; nt < NT; ++nt) {
      const __bf16* brow = wT + (nt * 16 + l15) * ldB + ko + hi * 8;
      bf16x16 b = ld_frag16(brow, brow + 16);
      acc[nt] = __builtin_amdgcn_wmma_f32_16x16x32_bf16(
          false, a, false, b, (short)0, acc[nt], false, false);
    }
  }
}

// C/D layout: lane = column (l&15), VGPR r -> row r (lanes 0-15) / r+8 (16-31).
template <int NT, bool RELU>
DEV void epilogue_bf16(const f32x8* acc, const float* __restrict__ bias,
                       __bf16* out, int ldO, int mRow, int lane) {
  const int l15  = lane & 15;
  const int rofs = ((lane >> 4) & 1) * 8;
#pragma unroll
  for (int nt = 0; nt < NT; ++nt) {
    const int   n  = nt * 16 + l15;
    const float bv = bias[n];
#pragma unroll
    for (int r = 0; r < 8; ++r) {
      float v = acc[nt][r] + bv;
      if (RELU) v = v > 0.0f ? v : 0.0f;
      out[(mRow + rofs + r) * ldO + n] = (__bf16)v;
    }
  }
}

template <int NT>
DEV void epilogue_f32(const f32x8* acc, const float* __restrict__ bias,
                      float* out, int ldO, int mRow, int lane) {
  const int l15  = lane & 15;
  const int rofs = ((lane >> 4) & 1) * 8;
#pragma unroll
  for (int nt = 0; nt < NT; ++nt) {
    const int   n  = nt * 16 + l15;
    const float bv = bias[n];
#pragma unroll
    for (int r = 0; r < 8; ++r)
      out[(mRow + rofs + r) * ldO + n] = acc[nt][r] + bv;
  }
}

// Stage W [K,N] (fp32, row-major, global) into LDS transposed [N,K] bf16.
DEV void stage_wT(const float* __restrict__ W, __bf16* wT, int K, int N,
                  int ldB, int tid) {
  for (int i = tid; i < K * N; i += NTHREADS) {
    const int k = i / N;
    const int n = i - k * N;              // coalesced global read across n
    wT[n * ldB + k] = (__bf16)W[i];
  }
}

struct MlpW {
  const float *W0, *b0, *W1, *b1, *W2, *b2;
};

// ---------------------------------------------------------------------------
// Edge MLP:  e_out = MLP(cat(x[row], x[col], e_in))   +  scatter-add by col
// ---------------------------------------------------------------------------
__global__ __launch_bounds__(NTHREADS)
void edge_mlp_kernel(const float* __restrict__ x,
                     const float* __restrict__ edge_attr,
                     const int*   __restrict__ eidx,     // [2*E] int32
                     MlpW w,
                     float* __restrict__ edge_out,
                     float* __restrict__ agg,
                     float* __restrict__ cnt) {
  constexpr int K1   = 3 * D;                    // 192
  constexpr int SZ_A = TM * LD_E1 * 2;           // 51200 B
  constexpr int SZ_W = TM * LD_E1 * 2;           // 51200 B (max 128x(192+8))
  constexpr int SZ_H = TM * LD_128 * 2;          // 34816 B  (>= 128*64*4)
  __shared__ __align__(16) unsigned char smem[SZ_A + SZ_W + SZ_H + 2 * TM * 4];

  __bf16* actIn = (__bf16*)smem;                         // [128, ld 200]
  __bf16* wT    = (__bf16*)(smem + SZ_A);                // weights^T
  __bf16* actH  = (__bf16*)(smem + SZ_A + SZ_W);         // [128, ld 136]
  float*  outF  = (float*)actH;                          // [128, 64] alias
  int* rowIds   = (int*)(smem + SZ_A + SZ_W + SZ_H);
  int* colIds   = rowIds + TM;

  const int  tid  = threadIdx.x;
  const int  lane = tid & 31;
  const int  wave = tid >> 5;
  const int  mRow = wave * 16;
  const long base = (long)blockIdx.x * TM;

  if (tid < TM) {
    const long e = base + tid;
    const int  r = eidx[e];
    const int  c = eidx[(long)EDGES + e];
    rowIds[tid] = r;
    colIds[tid] = c;
    atomic_add_f32(cnt + c, 1.0f);                 // degree count, once per edge
  }
  __syncthreads();

  // Gather cat(x[row], x[col], edge_attr) -> bf16 LDS tile [128,192]
  for (int i = tid; i < TM * 48; i += NTHREADS) {  // 48 float4 per row
    const int e  = i / 48;
    const int c4 = i - e * 48;
    f32x4 v;
    if (c4 < 16)
      v = ((const f32x4*)(x + (size_t)rowIds[e] * D))[c4];
    else if (c4 < 32)
      v = ((const f32x4*)(x + (size_t)colIds[e] * D))[c4 - 16];
    else
      v = ((const f32x4*)(edge_attr + (base + e) * D))[c4 - 32];
    bf16x4 bv = {(__bf16)v[0], (__bf16)v[1], (__bf16)v[2], (__bf16)v[3]};
    *(bf16x4*)(actIn + e * LD_E1 + c4 * 4) = bv;
  }
  stage_wT(w.W0, wT, K1, 128, LD_E1, tid);
  __syncthreads();

  f32x8 acc[8];
  wave_gemm<K1, 8>(actIn, LD_E1, wT, LD_E1, mRow, lane, acc);   // 192 -> 128
  epilogue_bf16<8, true>(acc, w.b0, actH, LD_128, mRow, lane);
  __syncthreads();

  stage_wT(w.W1, wT, 128, 128, LD_128, tid);
  __syncthreads();
  wave_gemm<128, 8>(actH, LD_128, wT, LD_128, mRow, lane, acc); // 128 -> 128
  epilogue_bf16<8, true>(acc, w.b1, actIn, LD_128, mRow, lane); // h2 -> actIn
  __syncthreads();

  stage_wT(w.W2, wT, 128, D, LD_128, tid);
  __syncthreads();
  wave_gemm<128, 4>(actIn, LD_128, wT, LD_128, mRow, lane, acc); // 128 -> 64
  epilogue_f32<4>(acc, w.b2, outF, D, mRow, lane);
  __syncthreads();

  // Coalesced store of new edge features
  for (int i = tid; i < TM * 16; i += NTHREADS) {
    const int e  = i >> 4;
    const int c4 = i & 15;
    ((f32x4*)(edge_out + (base + e) * D))[c4] = ((const f32x4*)(outF + e * D))[c4];
  }
  // Fused scatter-add into node accumulator (2 threads per edge, contiguous)
  {
    const int e    = tid >> 1;
    const int half = tid & 1;
    float*       dst = agg + (size_t)colIds[e] * D + half * 32;
    const float* src = outF + e * D + half * 32;
#pragma unroll
    for (int j = 0; j < 32; ++j) atomic_add_f32(dst + j, src[j]);
  }
}

// ---------------------------------------------------------------------------
// Node MLP:  x_out = MLP(cat(x, agg/max(cnt,1)))
// ---------------------------------------------------------------------------
__global__ __launch_bounds__(NTHREADS)
void node_mlp_kernel(const float* __restrict__ x,
                     const float* __restrict__ agg,
                     const float* __restrict__ cnt,
                     MlpW w,
                     float* __restrict__ x_out) {
  constexpr int SZ = TM * LD_128 * 2;                     // 34816 B each
  __shared__ __align__(16) unsigned char smem[3 * SZ];
  __bf16* actIn = (__bf16*)smem;
  __bf16* wT    = (__bf16*)(smem + SZ);
  __bf16* actH  = (__bf16*)(smem + 2 * SZ);
  float*  outF  = (float*)actH;                           // [128,64] alias

  const int tid  = threadIdx.x;
  const int lane = tid & 31;
  const int wave = tid >> 5;
  const int mRow = wave * 16;
  const int nb   = blockIdx.x * TM;

  // Build cat(x, mean) tile [128,128] bf16
  for (int i = tid; i < TM * 32; i += NTHREADS) {         // 32 float4 per row
    const int r  = i >> 5;
    const int c4 = i & 31;
    const int n  = nb + r;
    const int nc = n < NODES ? n : NODES - 1;
    f32x4 v;
    if (c4 < 16) {
      v = ((const f32x4*)(x + (size_t)nc * D))[c4];
    } else {
      float cd  = cnt[nc];
      cd        = cd > 1.0f ? cd : 1.0f;
      const float inv = 1.0f / cd;
      v = ((const f32x4*)(agg + (size_t)nc * D))[c4 - 16];
      v[0] *= inv; v[1] *= inv; v[2] *= inv; v[3] *= inv;
    }
    bf16x4 bv = {(__bf16)v[0], (__bf16)v[1], (__bf16)v[2], (__bf16)v[3]};
    *(bf16x4*)(actIn + r * LD_128 + c4 * 4) = bv;
  }
  stage_wT(w.W0, wT, 128, 128, LD_128, tid);
  __syncthreads();

  f32x8 acc[8];
  wave_gemm<128, 8>(actIn, LD_128, wT, LD_128, mRow, lane, acc);
  epilogue_bf16<8, true>(acc, w.b0, actH, LD_128, mRow, lane);
  __syncthreads();

  stage_wT(w.W1, wT, 128, 128, LD_128, tid);
  __syncthreads();
  wave_gemm<128, 8>(actH, LD_128, wT, LD_128, mRow, lane, acc);
  epilogue_bf16<8, true>(acc, w.b1, actIn, LD_128, mRow, lane);
  __syncthreads();

  stage_wT(w.W2, wT, 128, D, LD_128, tid);
  __syncthreads();
  wave_gemm<128, 4>(actIn, LD_128, wT, LD_128, mRow, lane, acc);
  epilogue_f32<4>(acc, w.b2, outF, D, mRow, lane);
  __syncthreads();

  for (int i = tid; i < TM * 16; i += NTHREADS) {
    const int r  = i >> 4;
    const int c4 = i & 15;
    const int n  = nb + r;
    if (n < NODES)
      ((f32x4*)(x_out + (size_t)n * D))[c4] = ((const f32x4*)(outF + r * D))[c4];
  }
}

__global__ void zero_kernel(float* __restrict__ p, long n) {
  const long i = (long)blockIdx.x * blockDim.x + threadIdx.x;
  if (i < n) p[i] = 0.0f;
}

extern "C" void kernel_launch(void* const* d_in, const int* in_sizes, int n_in,
                              void* d_out, int out_size, void* d_ws, size_t ws_size,
                              hipStream_t stream) {
  (void)in_sizes; (void)n_in; (void)out_size; (void)ws_size;
  const float* x0  = (const float*)d_in[0];
  const float* ea0 = (const float*)d_in[1];
  const int*   ei  = (const int*)d_in[2];

  auto mlp = [&](int b) {
    MlpW w;
    w.W0 = (const float*)d_in[b + 0]; w.b0 = (const float*)d_in[b + 1];
    w.W1 = (const float*)d_in[b + 2]; w.b1 = (const float*)d_in[b + 3];
    w.W2 = (const float*)d_in[b + 4]; w.b2 = (const float*)d_in[b + 5];
    return w;
  };
  // params flatten order: per layer {edge(W0,b0,W1,b1,W2,b2), node(...)}
  const MlpW e0 = mlp(3), n0 = mlp(9), e1 = mlp(15), n1 = mlp(21);

  // Workspace carve (fp32): agg[N*D] | cnt[N] | x1[N*D] | edge1[E*D]
  float* agg   = (float*)d_ws;
  float* cnt   = agg + (size_t)NODES * D;
  float* x1    = cnt + NODES;
  float* e1buf = x1 + (size_t)NODES * D;

  float* x_out = (float*)d_out;                 // [N, D]
  float* e_out = x_out + (size_t)NODES * D;     // [E, D]

  const long zn = (long)NODES * D + NODES;      // agg + cnt contiguous
  const int  zB = (int)((zn + 255) / 256);
  const int  eB = EDGES / TM;                   // 6250 exact
  const int  nB = (NODES + TM - 1) / TM;        // 391

  // ---- layer 0 ----
  zero_kernel<<<zB, 256, 0, stream>>>(agg, zn);
  edge_mlp_kernel<<<eB, NTHREADS, 0, stream>>>(x0, ea0, ei, e0, e1buf, agg, cnt);
  node_mlp_kernel<<<nB, NTHREADS, 0, stream>>>(x0, agg, cnt, n0, x1);
  // ---- layer 1 (final outputs straight into d_out) ----
  zero_kernel<<<zB, 256, 0, stream>>>(agg, zn);
  edge_mlp_kernel<<<eB, NTHREADS, 0, stream>>>(x1, e1buf, ei, e1, e_out, agg, cnt);
  node_mlp_kernel<<<nB, NTHREADS, 0, stream>>>(x1, agg, cnt, n1, x_out);
}
